// GraphConvolution_63883343560836
// MI455X (gfx1250) — matrix operand order repaired and checked by reading, not compiled, not run on hardware
//
#include <hip/hip_runtime.h>
#include <math.h>

// Problem constants (from reference)
#define NN 2048   // nodes
#define FF 128    // in features
#define OO 128    // out features
#define EE 16     // edge feature dim

typedef __attribute__((ext_vector_type(16))) __bf16 v16bf;
typedef __attribute__((ext_vector_type(8)))  __bf16 v8bf;
typedef __attribute__((ext_vector_type(8)))  float  v8f;
typedef __attribute__((ext_vector_type(4)))  float  v4f;

__device__ __forceinline__ __bf16 f2bf(float f) { return (__bf16)f; }

// A-fragment (16x32 bf16, row-major source): lane holds M = lane%16,
// K runs {ka..ka+7} and {ka+16..ka+23}, ka = k0 + (lane/16)*8.
__device__ __forceinline__ v16bf load_afrag(const __bf16* __restrict__ row, int ka) {
    v8bf lo = *(const v8bf*)(row + ka);
    v8bf hi = *(const v8bf*)(row + ka + 16);
    v16bf r;
    #pragma unroll
    for (int i = 0; i < 8; ++i) { r[i] = lo[i]; r[8 + i] = hi[i]; }
    return r;
}

// B-fragment (32x16 bf16, transposed [N][K] source): lane holds N = lane%16,
// K runs {kb..kb+7} and {kb+16..kb+23}, kb = k0 + (lane/16)*8.
__device__ __forceinline__ v16bf load_bfrag(const __bf16* __restrict__ row, int kb) {
    v8bf lo = *(const v8bf*)(row + kb);
    v8bf hi = *(const v8bf*)(row + kb + 16);
    v16bf r;
    #pragma unroll
    for (int i = 0; i < 8; ++i) { r[i] = lo[i]; r[8 + i] = hi[i]; }
    return r;
}

// ---------------------------------------------------------------------------
// Kernel A: pre{0,1} = x @ w{0,1}, written TRANSPOSED as bf16 preT[o][node].
// One wave per 16x16 output tile; K=128 -> 4 x wmma_16x16x32_bf16.
// ---------------------------------------------------------------------------
__global__ __launch_bounds__(32)
void pre_gemm_kernel(const float* __restrict__ x,    // [NN,FF] f32
                     const float* __restrict__ w0,   // [FF,OO] f32
                     const float* __restrict__ w1,   // [FF,OO] f32
                     __bf16* __restrict__ preT0,     // [OO,NN] bf16
                     __bf16* __restrict__ preT1) {   // [OO,NN] bf16
    const int lane = threadIdx.x;
    const int Ml   = lane & 15;
    const int hi   = lane >> 4;
    const int m0   = blockIdx.x * 16;      // node-row tile
    const int n0   = blockIdx.y * 16;      // out-channel tile
    const float* w    = blockIdx.z ? w1    : w0;
    __bf16*      preT = blockIdx.z ? preT1 : preT0;

    v8f acc = {0.f,0.f,0.f,0.f,0.f,0.f,0.f,0.f};

    for (int k0 = 0; k0 < FF; k0 += 32) {
        // A fragment from x (f32): two contiguous runs of 8, convert to bf16
        const float* xr = x + (size_t)(m0 + Ml) * FF + k0 + hi * 8;
        v4f r0 = *(const v4f*)(xr);
        v4f r1 = *(const v4f*)(xr + 4);
        v4f r2 = *(const v4f*)(xr + 16);
        v4f r3 = *(const v4f*)(xr + 20);
        v16bf a;
        a[0]=f2bf(r0.x); a[1]=f2bf(r0.y); a[2]=f2bf(r0.z);  a[3]=f2bf(r0.w);
        a[4]=f2bf(r1.x); a[5]=f2bf(r1.y); a[6]=f2bf(r1.z);  a[7]=f2bf(r1.w);
        a[8]=f2bf(r2.x); a[9]=f2bf(r2.y); a[10]=f2bf(r2.z); a[11]=f2bf(r2.w);
        a[12]=f2bf(r3.x);a[13]=f2bf(r3.y);a[14]=f2bf(r3.z); a[15]=f2bf(r3.w);

        // B fragment from w (f32 row-major [K][O]): strided gather (w is 64KB, L2-hot)
        v16bf b;
        #pragma unroll
        for (int j = 0; j < 16; ++j)
            b[j] = f2bf(w[(size_t)(k0 + hi * 8 + j + (j >= 8 ? 8 : 0)) * OO + n0 + Ml]);

        acc = __builtin_amdgcn_wmma_f32_16x16x32_bf16(
                  false, a, false, b, (short)0, acc, false, false);
    }

    // D layout: VGPR v -> M = v + hi*8, N = Ml.  Store transposed bf16.
    #pragma unroll
    for (int v = 0; v < 8; ++v)
        preT[(size_t)(n0 + Ml) * NN + (m0 + v + hi * 8)] = f2bf(acc[v]);
}

// ---------------------------------------------------------------------------
// Kernel B: streaming pass over the 512 MB edge tensors (HBM roofline term).
//   a1b[n,m] = bf16( sigmoid(dot16(ef,we)+bias_edge) * support1 )
//   s0b[n,m] = bf16( support0[n,m] )            (one-time f32->bf16 convert)
// Single-use inputs use non-temporal loads so the bf16 working set (~17 MB)
// stays resident in the 192 MB L2 for the WMMA kernel.
// ---------------------------------------------------------------------------
__global__ __launch_bounds__(256)
void gate_kernel(const float* __restrict__ edge_feature, // [NN,NN,EE]
                 const float* __restrict__ weight_edge,  // [NN,NN,EE]
                 const float* __restrict__ bias_edge,    // [NN,NN]
                 const float* __restrict__ support0,     // [NN,NN]
                 const float* __restrict__ support1,     // [NN,NN]
                 __bf16* __restrict__ a1b,               // [NN,NN] bf16
                 __bf16* __restrict__ s0b) {             // [NN,NN] bf16
    size_t idx = (size_t)blockIdx.x * 256 + threadIdx.x; // one (n,m) pair
    const v4f* ef = (const v4f*)(edge_feature + idx * EE);
    const v4f* we = (const v4f*)(weight_edge  + idx * EE);
    float dot = 0.f;
    #pragma unroll
    for (int i = 0; i < 4; ++i) {
        v4f a = __builtin_nontemporal_load(ef + i);
        v4f b = __builtin_nontemporal_load(we + i);
        dot += a.x * b.x + a.y * b.y + a.z * b.z + a.w * b.w;
    }
    float be = __builtin_nontemporal_load(bias_edge + idx);
    float s1 = __builtin_nontemporal_load(support1 + idx);
    float s0 = __builtin_nontemporal_load(support0 + idx);
    float g  = 1.0f / (1.0f + __expf(-(dot + be)));
    a1b[idx] = f2bf(g * s1);
    s0b[idx] = f2bf(s0);
}

// ---------------------------------------------------------------------------
// Kernel C: out = relu(support0 @ pre0 + a1 @ pre1 + b)
// One wave per 32x16 output tile (two row-tiles share the B fragments ->
// 3 B128 loads per WMMA instead of 4). K=2048 -> 64 iters x 4 WMMA.
// All operands are bf16 and L2-resident; 512 waves in flight.
// ---------------------------------------------------------------------------
__global__ __launch_bounds__(32)
void agg_gemm_kernel(const __bf16* __restrict__ s0b,    // [NN,NN] bf16
                     const __bf16* __restrict__ a1b,    // [NN,NN] bf16
                     const __bf16* __restrict__ preT0,  // [OO,NN] bf16
                     const __bf16* __restrict__ preT1,  // [OO,NN] bf16
                     const float*  __restrict__ bias,   // [OO]
                     float* __restrict__ out) {         // [NN,OO] f32
    const int lane = threadIdx.x;
    const int Ml   = lane & 15;
    const int hi   = lane >> 4;
    const int m0   = blockIdx.x * 32;  // row-pair tile (64)
    const int n0   = blockIdx.y * 16;  // out-channel tile (8)

    const __bf16* s0r0 = s0b   + (size_t)(m0 + Ml) * NN;        // rows m0..m0+15
    const __bf16* s0r1 = s0r0  + (size_t)16 * NN;               // rows m0+16..m0+31
    const __bf16* a1r0 = a1b   + (size_t)(m0 + Ml) * NN;
    const __bf16* a1r1 = a1r0  + (size_t)16 * NN;
    const __bf16* b0row = preT0 + (size_t)(n0 + Ml) * NN;
    const __bf16* b1row = preT1 + (size_t)(n0 + Ml) * NN;

    v8f acc0 = {0.f,0.f,0.f,0.f,0.f,0.f,0.f,0.f};
    v8f acc1 = {0.f,0.f,0.f,0.f,0.f,0.f,0.f,0.f};

    for (int k0 = 0; k0 < NN; k0 += 32) {
        const int ka = k0 + hi * 8;
        const int kb = k0 + hi * 8;

        v16bf A0r0 = load_afrag(s0r0, ka);
        v16bf A0r1 = load_afrag(s0r1, ka);
        v16bf A1r0 = load_afrag(a1r0, ka);
        v16bf A1r1 = load_afrag(a1r1, ka);
        v16bf B0   = load_bfrag(b0row, kb);
        v16bf B1   = load_bfrag(b1row, kb);

        acc0 = __builtin_amdgcn_wmma_f32_16x16x32_bf16(
                   false, A0r0, false, B0, (short)0, acc0, false, false);
        acc1 = __builtin_amdgcn_wmma_f32_16x16x32_bf16(
                   false, A0r1, false, B0, (short)0, acc1, false, false);
        acc0 = __builtin_amdgcn_wmma_f32_16x16x32_bf16(
                   false, A1r0, false, B1, (short)0, acc0, false, false);
        acc1 = __builtin_amdgcn_wmma_f32_16x16x32_bf16(
                   false, A1r1, false, B1, (short)0, acc1, false, false);
    }

    const float bv = bias[n0 + Ml];
    #pragma unroll
    for (int v = 0; v < 8; ++v) {
        const int Mm = v + hi * 8;
        float r0 = acc0[v] + bv;
        float r1 = acc1[v] + bv;
        out[(size_t)(m0 + Mm)      * OO + (n0 + Ml)] = fmaxf(r0, 0.f);
        out[(size_t)(m0 + 16 + Mm) * OO + (n0 + Ml)] = fmaxf(r1, 0.f);
    }
}

// ---------------------------------------------------------------------------
// Launcher. Workspace layout (17 MB, all bf16):
//   [0, 512KB)      preT0  [OO][NN]
//   [512KB, 1MB)    preT1  [OO][NN]
//   [1MB, 9MB)      a1b    [NN][NN]   gated adjacency
//   [9MB, 17MB)     s0b    [NN][NN]   support0 as bf16
// ---------------------------------------------------------------------------
extern "C" void kernel_launch(void* const* d_in, const int* in_sizes, int n_in,
                              void* d_out, int out_size, void* d_ws, size_t ws_size,
                              hipStream_t stream) {
    const float* x       = (const float*)d_in[0];  // [NN,FF]
    const float* support = (const float*)d_in[1];  // [2,NN,NN]
    const float* ef      = (const float*)d_in[2];  // [NN,NN,EE]
    const float* we      = (const float*)d_in[3];  // [NN,NN,EE]
    const float* be      = (const float*)d_in[4];  // [NN,NN]
    const float* w0      = (const float*)d_in[5];  // [FF,OO]
    const float* w1      = (const float*)d_in[6];  // [FF,OO]
    const float* b       = (const float*)d_in[7];  // [OO]
    float* out           = (float*)d_out;          // [NN,OO]

    char* ws = (char*)d_ws;
    __bf16* preT0 = (__bf16*)(ws);
    __bf16* preT1 = (__bf16*)(ws + (size_t)OO * NN * 2);
    __bf16* a1b   = (__bf16*)(ws + (size_t)OO * NN * 4);
    __bf16* s0b   = (__bf16*)(ws + (size_t)OO * NN * 4 + (size_t)NN * NN * 2);

    const float* support0 = support;
    const float* support1 = support + (size_t)NN * NN;

    // A: pre-GEMMs (both weights), transposed bf16 outputs
    pre_gemm_kernel<<<dim3(NN / 16, OO / 16, 2), 32, 0, stream>>>(
        x, w0, w1, preT0, preT1);

    // B: streaming edge-gate pass + one-time support0 bf16 convert
    gate_kernel<<<dim3((NN * NN) / 256), 256, 0, stream>>>(
        ef, we, be, support0, support1, a1b, s0b);

    // C: fused dual-branch aggregation GEMM + bias + relu
    agg_gemm_kernel<<<dim3(NN / 32, OO / 16), 32, 0, stream>>>(
        s0b, a1b, preT0, preT1, b, out);
}
